// Attention_50216757625638
// MI455X (gfx1250) — matrix operand order repaired and checked by reading, not compiled, not run on hardware
//
#include <hip/hip_runtime.h>
#include <hip/hip_bf16.h>

typedef __attribute__((ext_vector_type(16))) __bf16 v16bf;
typedef __attribute__((ext_vector_type(8)))  float  v8f;

// ---------- helpers ----------

__device__ __forceinline__ unsigned short f2bf(float f) {
    unsigned int u = __float_as_uint(f);
    u = (u + 0x7FFFu + ((u >> 16) & 1u)) >> 16;   // RNE
    return (unsigned short)u;
}

// load a v16bf fragment as two 16B chunks (ds_load_b128 / global_load_b128)
__device__ __forceinline__ v16bf ldfrag(const unsigned short* p0, const unsigned short* p1) {
    union { v16bf v; uint4 u[2]; } f;
    f.u[0] = *(const uint4*)p0;
    f.u[1] = *(const uint4*)p1;
    return f.v;
}

__device__ __forceinline__ v8f wmma_bf16(v16bf a, v16bf b, v8f c) {
    return __builtin_amdgcn_wmma_f32_16x16x32_bf16(false, a, false, b, (short)0, c, false, false);
}

__device__ __forceinline__ v8f vzero8() {
    v8f z;
#pragma unroll
    for (int i = 0; i < 8; ++i) z[i] = 0.0f;
    return z;
}

// CDNA5 async global->LDS copy (16B per lane), tracked by ASYNCcnt.
// vdst = LDS byte offset, vaddr = 64-bit global address, "off" = no SADDR.
__device__ __forceinline__ void async_b128(const void* gsrc, unsigned ldsByteOff) {
    asm volatile("global_load_async_to_lds_b128 %0, %1, off"
                 :: "v"(ldsByteOff), "v"((unsigned long long)(size_t)gsrc)
                 : "memory");
}
__device__ __forceinline__ void wait_async0() {
    asm volatile("s_wait_asynccnt 0" ::: "memory");
}
// steady-state double-buffer wait: tolerate the 4 newest (next tile) in flight
__device__ __forceinline__ void wait_async4() {
    asm volatile("s_wait_asynccnt 4" ::: "memory");
}
__device__ __forceinline__ unsigned lds_off(const void* p) {
    return (unsigned)(size_t)p;   // generic LDS pointer: low 32 bits = LDS byte offset
}

// ---------- pre-pass: fp32 -> bf16 (plain and transposed) ----------

__global__ __launch_bounds__(256) void cvt_bf16(const float* __restrict__ in,
                                                unsigned short* __restrict__ out, int n) {
    int i = blockIdx.x * 256 + threadIdx.x;
    if (i < n) out[i] = f2bf(in[i]);
}

// in [K][N] row-major  ->  out [N][K] row-major (W^T)
__global__ __launch_bounds__(256) void cvt_bf16_T(const float* __restrict__ in,
                                                  unsigned short* __restrict__ out,
                                                  int K, int N) {
    int i = blockIdx.x * 256 + threadIdx.x;
    if (i < K * N) {
        int k = i / N, n = i - k * N;
        out[(size_t)n * K + k] = f2bf(in[i]);
    }
}

// ---------- GEMM: C[M,N] = A[M,K] * Bt[N,K]^T + bias[N]  (bf16 in, WMMA, f32 acc)
// WG = 256 threads = 8 waves; 128x128 WG tile; 32x64 wave tile.
// Double-buffered LDS: async DMA of tile t+1 overlaps WMMA on tile t.

template <bool OUT_BF16>
__global__ __launch_bounds__(256) void gemm_bias_wmma(
    const unsigned short* __restrict__ A,   // [M,K] bf16
    const unsigned short* __restrict__ Bt,  // [N,K] bf16 (pre-transposed weights)
    const float* __restrict__ bias,
    void* __restrict__ Cout,
    int M, int N, int K)
{
    __shared__ __attribute__((aligned(16))) unsigned short As[2][128 * 32];  // [m][k]
    __shared__ __attribute__((aligned(16))) unsigned short Bs[2][128 * 32];  // [n][k]

    const int tid   = threadIdx.x;
    const int lane  = tid & 31;
    const int wave  = tid >> 5;
    const int laneN = lane & 15;
    const int laneH = lane >> 4;

    const int nb = N >> 7;
    const int m0 = (blockIdx.x / nb) << 7;
    const int n0 = (blockIdx.x % nb) << 7;
    const int wm = wave >> 1;
    const int wn = wave & 1;

    const unsigned asBase = lds_off(As);
    const unsigned bsBase = lds_off(Bs);

    // stage k-tile kt into buffer buf: 4 async_b128 per wave
    auto stage = [&](int kt, int buf) {
        const int k0 = kt * 32;
        const unsigned aB = asBase + (unsigned)buf * (128 * 32 * 2);
        const unsigned bB = bsBase + (unsigned)buf * (128 * 32 * 2);
        for (int i = tid; i < 512; i += 256) {
            int r = i >> 2, c = (i & 3) * 8;
            async_b128(&A [(size_t)(m0 + r) * K + k0 + c], aB + (unsigned)(r * 32 + c) * 2);
            async_b128(&Bt[(size_t)(n0 + r) * K + k0 + c], bB + (unsigned)(r * 32 + c) * 2);
        }
    };

    v8f acc[2][4];
#pragma unroll
    for (int i = 0; i < 2; ++i)
#pragma unroll
        for (int j = 0; j < 4; ++j) acc[i][j] = vzero8();

    const int NT = K >> 5;
    stage(0, 0);   // prologue

    for (int t = 0; t < NT; ++t) {
        const int buf = t & 1;
        if (t + 1 < NT) {
            stage(t + 1, buf ^ 1);        // DMA next tile into the other buffer
            wait_async4();                // wait only for tile t's 4 older asyncs
        } else {
            wait_async0();
        }
        if (t + 2 < NT) {                 // L2 prefetch two tiles ahead
            __builtin_prefetch(&A [(size_t)(m0 + (tid >> 1)) * K + (t + 2) * 32 + (tid & 1) * 16], 0, 1);
            __builtin_prefetch(&Bt[(size_t)(n0 + (tid >> 1)) * K + (t + 2) * 32 + (tid & 1) * 16], 0, 1);
        }
        __syncthreads();

        const unsigned short* Ab = As[buf];
        const unsigned short* Bb = Bs[buf];
        v16bf af[2], bfm[4];
#pragma unroll
        for (int mt = 0; mt < 2; ++mt) {
            const unsigned short* p = &Ab[(wm * 32 + mt * 16 + laneN) * 32 + laneH * 8];
            af[mt] = ldfrag(p, p + 16);
        }
#pragma unroll
        for (int nt = 0; nt < 4; ++nt) {
            const unsigned short* p = &Bb[(wn * 64 + nt * 16 + laneN) * 32 + laneH * 8];
            bfm[nt] = ldfrag(p, p + 16);
        }
#pragma unroll
        for (int mt = 0; mt < 2; ++mt)
#pragma unroll
            for (int nt = 0; nt < 4; ++nt)
                acc[mt][nt] = wmma_bf16(af[mt], bfm[nt], acc[mt][nt]);
        __syncthreads();   // tile t fully consumed; its buffer may be overwritten
    }

#pragma unroll
    for (int nt = 0; nt < 4; ++nt) {
        int col = n0 + wn * 64 + nt * 16 + laneN;
        float bv = bias[col];
#pragma unroll
        for (int mt = 0; mt < 2; ++mt) {
#pragma unroll
            for (int r = 0; r < 8; ++r) {
                int row = m0 + wm * 32 + mt * 16 + r + 8 * laneH;
                float v = acc[mt][nt][r] + bv;
                if (OUT_BF16) ((unsigned short*)Cout)[(size_t)row * N + col] = f2bf(v);
                else          ((float*)Cout)[(size_t)row * N + col] = v;
            }
        }
    }
}

// ---------- Flash attention (causal), bf16 qkv in, bf16 out ----------
// grid: (B*H, T/128). WG = 256 threads = 8 waves; wave owns 16 query rows.

__global__ __launch_bounds__(256) void flash_attn_bf16(
    const unsigned short* __restrict__ qkv,   // [B,T,3C] bf16
    unsigned short* __restrict__ outp)        // [B,T,C]  bf16
{
    constexpr int Tq = 2048, Cc = 1024, Hh = 16, Dd = 64;
    __shared__ __attribute__((aligned(16))) unsigned short Ks[128 * 64];      // [key][d]
    __shared__ __attribute__((aligned(16))) unsigned short Vs[64 * 128];      // [d][key]
    __shared__ __attribute__((aligned(16))) unsigned short Ps[8 * 16 * 128];  // per-wave P

    const int tid   = threadIdx.x;
    const int lane  = tid & 31;
    const int wave  = tid >> 5;
    const int laneN = lane & 15;
    const int laneH = lane >> 4;

    const int bh = blockIdx.x;
    const int bb = bh / Hh;
    const int h  = bh % Hh;
    const int qb = blockIdx.y;
    const int q0 = qb * 128;

    // Q A-fragments straight from global bf16
    const int q_row = q0 + wave * 16 + laneN;
    const unsigned short* qp = qkv + ((size_t)bb * Tq + q_row) * (3 * Cc) + h * Dd;
    v16bf qf[2];
#pragma unroll
    for (int kk = 0; kk < 2; ++kk) {
        int bd = kk * 32 + laneH * 8;
        qf[kk] = ldfrag(qp + bd, qp + bd + 16);
    }

    v8f o[4];
#pragma unroll
    for (int dt = 0; dt < 4; ++dt) o[dt] = vzero8();
    float m[8], l[8];
#pragma unroll
    for (int r = 0; r < 8; ++r) { m[r] = -__builtin_inff(); l[r] = 0.0f; }

    const float CL2 = 0.125f * 1.44269504088896340736f;   // D^-1/2 * log2(e)
    const unsigned ksBase = lds_off(Ks);

    for (int kb = 0; kb <= qb; ++kb) {
        const size_t rowB = ((size_t)bb * Tq + kb * 128) * (3 * Cc) + h * Dd;
        // K tile: async DMA, 1024 x 16B chunks
        for (int i = tid; i < 1024; i += 256) {
            int r = i >> 3, c = (i & 7) * 8;
            async_b128(&qkv[rowB + Cc + (size_t)r * (3 * Cc) + c],
                       ksBase + (unsigned)(r * 64 + c) * 2);
        }
        // V tile: transpose through VGPRs (b128 load + 8x ds_store_b16)
        for (int i = tid; i < 1024; i += 256) {
            int r = i >> 3, c = (i & 7) * 8;
            union { uint4 u; unsigned short e[8]; } w;
            w.u = *(const uint4*)&qkv[rowB + 2 * Cc + (size_t)r * (3 * Cc) + c];
#pragma unroll
            for (int j = 0; j < 8; ++j) Vs[(c + j) * 128 + r] = w.e[j];
        }
        wait_async0();
        __syncthreads();

        // S = Q K^T
        v8f s[8];
#pragma unroll
        for (int n = 0; n < 8; ++n) {
            s[n] = vzero8();
            const unsigned short* kp = &Ks[(n * 16 + laneN) * 64 + laneH * 8];
            s[n] = wmma_bf16(qf[0], ldfrag(kp,      kp + 16), s[n]);
            s[n] = wmma_bf16(qf[1], ldfrag(kp + 32, kp + 48), s[n]);
        }

        if (kb == qb) {
#pragma unroll
            for (int n = 0; n < 8; ++n) {
                int col = kb * 128 + n * 16 + laneN;
#pragma unroll
                for (int r = 0; r < 8; ++r) {
                    int rowq = q0 + wave * 16 + r + 8 * laneH;
                    if (col > rowq) s[n][r] = -__builtin_inff();
                }
            }
        }

        // online softmax (row M = r + 8*laneH, replicated over each 16-lane half)
        unsigned short* pw = &Ps[wave * 2048];
#pragma unroll
        for (int r = 0; r < 8; ++r) {
            float mx = m[r];
#pragma unroll
            for (int n = 0; n < 8; ++n) mx = fmaxf(mx, s[n][r]);
            mx = fmaxf(mx, __shfl_xor(mx, 1, 32));
            mx = fmaxf(mx, __shfl_xor(mx, 2, 32));
            mx = fmaxf(mx, __shfl_xor(mx, 4, 32));
            mx = fmaxf(mx, __shfl_xor(mx, 8, 32));
            float corr = exp2f((m[r] - mx) * CL2);
            m[r] = mx;
            float rs = 0.0f;
#pragma unroll
            for (int n = 0; n < 8; ++n) {
                float p = exp2f((s[n][r] - mx) * CL2);
                rs += p;
                pw[(r + 8 * laneH) * 128 + n * 16 + laneN] = f2bf(p);
            }
            rs += __shfl_xor(rs, 1, 32);
            rs += __shfl_xor(rs, 2, 32);
            rs += __shfl_xor(rs, 4, 32);
            rs += __shfl_xor(rs, 8, 32);
            l[r] = l[r] * corr + rs;
#pragma unroll
            for (int dt = 0; dt < 4; ++dt) o[dt][r] *= corr;
        }

        // O += P V (same-wave LDS round-trip, DS ops are in-order per wave)
#pragma unroll
        for (int kc = 0; kc < 4; ++kc) {
            const unsigned short* pp = &pw[laneN * 128 + kc * 32 + laneH * 8];
            v16bf pf = ldfrag(pp, pp + 16);
#pragma unroll
            for (int dt = 0; dt < 4; ++dt) {
                const unsigned short* vp = &Vs[(dt * 16 + laneN) * 128 + kc * 32 + laneH * 8];
                o[dt] = wmma_bf16(pf, ldfrag(vp, vp + 16), o[dt]);
            }
        }
        __syncthreads();
    }

    // normalize, convert once to bf16, store out[b, t, h*D + d]
#pragma unroll
    for (int r = 0; r < 8; ++r) {
        float invl = 1.0f / l[r];
        int row = q0 + wave * 16 + r + 8 * laneH;
#pragma unroll
        for (int dt = 0; dt < 4; ++dt) {
            int d = dt * 16 + laneN;
            outp[((size_t)bb * Tq + row) * Cc + h * Dd + d] = f2bf(o[dt][r] * invl);
        }
    }
}

// ---------- launcher ----------

extern "C" void kernel_launch(void* const* d_in, const int* in_sizes, int n_in,
                              void* d_out, int out_size, void* d_ws, size_t ws_size,
                              hipStream_t stream) {
    const float* x     = (const float*)d_in[0];   // [4,2048,1024]
    const float* W_qkv = (const float*)d_in[1];   // [1024,3072]
    const float* b_qkv = (const float*)d_in[2];   // [3072]
    const float* W_o   = (const float*)d_in[3];   // [1024,1024]
    const float* b_o   = (const float*)d_in[4];   // [1024]
    float* out = (float*)d_out;

    const int M = 4 * 2048;   // 8192
    const int C = 1024;

    // bf16 workspace layout (~92 MB total, L2-resident working set)
    unsigned short* xb   = (unsigned short*)d_ws;            // [8192,1024]
    unsigned short* Wqt  = xb   + (size_t)M * C;             // [3072,1024] = W_qkv^T
    unsigned short* Wot  = Wqt  + (size_t)3 * C * C;         // [1024,1024] = W_o^T
    unsigned short* qkvb = Wot  + (size_t)C * C;             // [8192,3072]
    unsigned short* attb = qkvb + (size_t)M * 3 * C;         // [8192,1024]

    dim3 blk(256);

    // 0) one-time precision conversion / weight transpose
    cvt_bf16  <<<dim3((M * C) / 256), blk, 0, stream>>>(x, xb, M * C);
    cvt_bf16_T<<<dim3((3 * C * C + 255) / 256), blk, 0, stream>>>(W_qkv, Wqt, C, 3 * C);
    cvt_bf16_T<<<dim3((C * C + 255) / 256), blk, 0, stream>>>(W_o, Wot, C, C);

    // 1) qkv = x @ W_qkv + b_qkv   (bf16 out)
    gemm_bias_wmma<true><<<dim3((M / 128) * (3 * C / 128)), blk, 0, stream>>>(
        xb, Wqt, b_qkv, qkvb, M, 3 * C, C);

    // 2) causal flash attention (bf16 in/out)
    flash_attn_bf16<<<dim3(4 * 16, 2048 / 128), blk, 0, stream>>>(qkvb, attb);

    // 3) out = attn @ W_o + b_o    (f32 out)
    gemm_bias_wmma<false><<<dim3((M / 128) * (C / 128)), blk, 0, stream>>>(
        attb, Wot, b_o, out, M, C, C);
}
